// MultiheadAttention_89893665505280
// MI455X (gfx1250) — matrix operand order repaired and checked by reading, not compiled, not run on hardware
//
#include <hip/hip_runtime.h>
#include <hip/hip_bf16.h>

typedef __attribute__((ext_vector_type(16))) _Float16 v16h;
typedef __attribute__((ext_vector_type(8)))  float    v8f;

#define D_MODEL 256
#define NHEAD   8
#define DK      32
#define T_LEN   1024
#define S_LEN   1024
#define BATCH   8
#define MROWS   (T_LEN * BATCH)          // 8192 rows for all projections
#define SCALE   0.17677669529663687f     // 1/sqrt(32)

// ---------------------------------------------------------------------------
// Positional-bias precompute: pq[b,t] = dot(query_xyz[t,b,:], pe_w)
//                             pk[b,s] = dot(key_xyz  [s,b,:], pe_w)
// ---------------------------------------------------------------------------
__global__ void pe_dot_kernel(const float* __restrict__ qx, const float* __restrict__ kx,
                              const float* __restrict__ pe_w,
                              float* __restrict__ pq, float* __restrict__ pk) {
  int tid = blockIdx.x * blockDim.x + threadIdx.x;
  float w0 = pe_w[0], w1 = pe_w[1], w2 = pe_w[2];
  if (tid < T_LEN * BATCH) {
    int t = tid >> 3, b = tid & 7;
    const float* p = qx + (size_t)tid * 3;
    pq[b * T_LEN + t] = p[0] * w0 + p[1] * w1 + p[2] * w2;
  } else if (tid < 2 * T_LEN * BATCH) {
    int i = tid - T_LEN * BATCH;
    int s = i >> 3, b = i & 7;
    const float* p = kx + (size_t)i * 3;
    pk[b * S_LEN + s] = p[0] * w0 + p[1] * w1 + p[2] * w2;
  }
}

// ---------------------------------------------------------------------------
// Y[m,n] = sum_k X[m,k] * W[n,k] + bias[n]   (X f32, Y f16)  M=8192, N=K=256
// One wave per 16x16 tile. WMMA f16 16x16x32, f32 accumulate.
// ---------------------------------------------------------------------------
__global__ void __launch_bounds__(32) gemm_f32_to_f16(
    const float* __restrict__ X, const float* __restrict__ W,
    const float* __restrict__ bias, _Float16* __restrict__ Y) {
  const int lane = threadIdx.x & 31;
  const int col  = lane & 15;     // A: row index / B: column index
  const int hi   = lane >> 4;     // lane-half select
  const int m0 = blockIdx.x * 16;
  const int n0 = blockIdx.y * 16;
  v8f acc = {};
  const float* xr = X + (size_t)(m0 + col) * D_MODEL;
  const float* wr = W + (size_t)(n0 + col) * D_MODEL;
#pragma unroll
  for (int k0 = 0; k0 < D_MODEL; k0 += 32) {
    v16h a, bf;
#pragma unroll
    for (int e = 0; e < 8; ++e) {
      a[e]     = (_Float16)xr[k0 + hi * 16 + e];
      a[8 + e] = (_Float16)xr[k0 + hi * 16 + 8 + e];
    }
#pragma unroll
    for (int e = 0; e < 16; ++e) bf[e] = (_Float16)wr[k0 + hi * 16 + e];
    acc = __builtin_amdgcn_wmma_f32_16x16x32_f16(false, a, false, bf, (short)0, acc, false, false);
  }
  const float bn = bias[n0 + col];
#pragma unroll
  for (int r = 0; r < 8; ++r)
    Y[(size_t)(m0 + r + 8 * hi) * D_MODEL + n0 + col] = (_Float16)(acc[r] + bn);
}

// V projection: same GEMM but stores K-major transposed layout
//   Vt[((b*NHEAD + h)*DK + d) * S_LEN + s]   where m = s*BATCH+b, n = h*DK+d
// so pass-2's P·V B-fragments become contiguous b128 loads along S.
__global__ void __launch_bounds__(32) gemm_f32_to_f16_vt(
    const float* __restrict__ X, const float* __restrict__ W,
    const float* __restrict__ bias, _Float16* __restrict__ Vt) {
  const int lane = threadIdx.x & 31;
  const int col = lane & 15, hi = lane >> 4;
  const int m0 = blockIdx.x * 16, n0 = blockIdx.y * 16;
  v8f acc = {};
  const float* xr = X + (size_t)(m0 + col) * D_MODEL;
  const float* wr = W + (size_t)(n0 + col) * D_MODEL;
#pragma unroll
  for (int k0 = 0; k0 < D_MODEL; k0 += 32) {
    v16h a, bf;
#pragma unroll
    for (int e = 0; e < 8; ++e) {
      a[e]     = (_Float16)xr[k0 + hi * 16 + e];
      a[8 + e] = (_Float16)xr[k0 + hi * 16 + 8 + e];
    }
#pragma unroll
    for (int e = 0; e < 16; ++e) bf[e] = (_Float16)wr[k0 + hi * 16 + e];
    acc = __builtin_amdgcn_wmma_f32_16x16x32_f16(false, a, false, bf, (short)0, acc, false, false);
  }
  const int n = n0 + col;
  const int hh = n >> 5, dd = n & 31;
  const float bn = bias[n];
#pragma unroll
  for (int r = 0; r < 8; ++r) {
    const int m = m0 + r + 8 * hi;
    const int s = m >> 3, bb = m & 7;
    Vt[(((size_t)(bb * NHEAD + hh)) * DK + dd) * S_LEN + s] = (_Float16)(acc[r] + bn);
  }
}

// Output projection: f16 input (ctx), f32 output straight into d_out
// ([T,B,D], row = t*B+b)
__global__ void __launch_bounds__(32) gemm_f16_to_f32(
    const _Float16* __restrict__ X, const float* __restrict__ W,
    const float* __restrict__ bias, float* __restrict__ Y) {
  const int lane = threadIdx.x & 31;
  const int col = lane & 15, hi = lane >> 4;
  const int m0 = blockIdx.x * 16, n0 = blockIdx.y * 16;
  v8f acc = {};
  const _Float16* xr = X + (size_t)(m0 + col) * D_MODEL;
  const float* wr = W + (size_t)(n0 + col) * D_MODEL;
#pragma unroll
  for (int k0 = 0; k0 < D_MODEL; k0 += 32) {
    v16h a, bf;
#pragma unroll
    for (int e = 0; e < 8; ++e) {
      a[e]     = xr[k0 + hi * 16 + e];
      a[8 + e] = xr[k0 + hi * 16 + 8 + e];
    }
#pragma unroll
    for (int e = 0; e < 16; ++e) bf[e] = (_Float16)wr[k0 + hi * 16 + e];
    acc = __builtin_amdgcn_wmma_f32_16x16x32_f16(false, a, false, bf, (short)0, acc, false, false);
  }
  const float bn = bias[n0 + col];
#pragma unroll
  for (int r = 0; r < 8; ++r)
    Y[(size_t)(m0 + r + 8 * hi) * D_MODEL + n0 + col] = acc[r] + bn;
}

// ---------------------------------------------------------------------------
// Pass 1: per (b,h,t-tile) stream over S, one 16x16x32 WMMA per 16 keys
// (d_k == 32 == one WMMA K-step). Online softmax stats (rowmax, rowsumexp).
// ---------------------------------------------------------------------------
__global__ void __launch_bounds__(32) attn_pass1(
    const _Float16* __restrict__ Qh, const _Float16* __restrict__ Kh,
    const float* __restrict__ pq, const float* __restrict__ pk,
    const float* __restrict__ pe_b,
    float* __restrict__ rowm, float* __restrict__ rowl) {
  const int lane = threadIdx.x & 31;
  const int col = lane & 15, hi = lane >> 4;
  const int t0 = blockIdx.x * 16;
  const int h  = blockIdx.y;
  const int b  = blockIdx.z;
  const float peb = pe_b[0];

  // A fragment: Q rows t0..t0+15, K-dim = d_k (full 32)
  const _Float16* qr = Qh + ((size_t)(t0 + col) * BATCH + b) * D_MODEL + h * DK;
  v16h a;
#pragma unroll
  for (int e = 0; e < 8; ++e) { a[e] = qr[hi * 16 + e]; a[8 + e] = qr[hi * 16 + 8 + e]; }

  float pqr[8], m[8], l[8];
#pragma unroll
  for (int r = 0; r < 8; ++r) {
    pqr[r] = pq[b * T_LEN + t0 + r + 8 * hi] + peb;
    m[r] = -1e30f; l[r] = 0.f;
  }

  for (int s0 = 0; s0 < S_LEN; s0 += 16) {
    // B fragment: K^T — column s = lane, 16 contiguous d-values per lane half
    const _Float16* kr = Kh + ((size_t)(s0 + col) * BATCH + b) * D_MODEL + h * DK + hi * 16;
    // prefetch 2 key-tiles ahead (gfx1250 global_prefetch_b8)
    __builtin_prefetch(kr + (size_t)2 * 16 * BATCH * D_MODEL, 0, 0);
    v16h bf;
#pragma unroll
    for (int e = 0; e < 16; ++e) bf[e] = kr[e];
    v8f c = {};
    c = __builtin_amdgcn_wmma_f32_16x16x32_f16(false, a, false, bf, (short)0, c, false, false);
    const float pks = pk[b * S_LEN + s0 + col];
#pragma unroll
    for (int r = 0; r < 8; ++r) {
      float v  = c[r] * SCALE + pqr[r] - pks;
      float nm = fmaxf(m[r], v);
      l[r] = l[r] * __expf(m[r] - nm) + __expf(v - nm);
      m[r] = nm;
    }
  }
  // merge the 16 lane-columns sharing each row (xor within 16-lane groups)
#pragma unroll
  for (int mask = 1; mask < 16; mask <<= 1) {
#pragma unroll
    for (int r = 0; r < 8; ++r) {
      float om = __shfl_xor(m[r], mask, 32);
      float ol = __shfl_xor(l[r], mask, 32);
      float nm = fmaxf(m[r], om);
      l[r] = l[r] * __expf(m[r] - nm) + ol * __expf(om - nm);
      m[r] = nm;
    }
  }
  if (col == 0) {
    size_t base = ((size_t)(b * NHEAD + h)) * T_LEN + t0 + 8 * hi;
#pragma unroll
    for (int r = 0; r < 8; ++r) { rowm[base + r] = m[r]; rowl[base + r] = l[r]; }
  }
}

// ---------------------------------------------------------------------------
// Pass 2: block = (b, t-tile), 8 waves = 8 heads. Recompute score tiles,
// finalize probs, LDS-atomic head-mean accumulation (chunked writeout),
// LDS-stage probs -> WMMA-A fragment, ctx += P·V (2 WMMAs / 32-key chunk).
// V read from the K-major transposed Vt buffer -> contiguous b128 fragments.
// ---------------------------------------------------------------------------
__global__ void __launch_bounds__(256) attn_pass2(
    const _Float16* __restrict__ Qh, const _Float16* __restrict__ Kh,
    const _Float16* __restrict__ Vt,
    const float* __restrict__ pq, const float* __restrict__ pk,
    const float* __restrict__ pe_b,
    const float* __restrict__ rowm, const float* __restrict__ rowl,
    float* __restrict__ meanbuf, _Float16* __restrict__ Ch) {
  __shared__ float    smean[16 * 32];            // head-mean tile chunk
  __shared__ _Float16 pstage[NHEAD][16 * 32];    // per-wave prob staging
  const int wave = threadIdx.x >> 5;   // = head
  const int lane = threadIdx.x & 31;
  const int col = lane & 15, hi = lane >> 4;
  const int t0 = blockIdx.x * 16;
  const int b  = blockIdx.y;
  const int h  = wave;
  const float peb = pe_b[0];

  const _Float16* qr = Qh + ((size_t)(t0 + col) * BATCH + b) * D_MODEL + h * DK;
  v16h a;
#pragma unroll
  for (int e = 0; e < 8; ++e) { a[e] = qr[hi * 16 + e]; a[8 + e] = qr[hi * 16 + 8 + e]; }

  float pqr[8], mrow[8], ilrow[8];
  {
    size_t base = ((size_t)(b * NHEAD + h)) * T_LEN + t0 + 8 * hi;
#pragma unroll
    for (int r = 0; r < 8; ++r) {
      pqr[r]   = pq[b * T_LEN + t0 + r + 8 * hi] + peb;
      mrow[r]  = rowm[base + r];
      ilrow[r] = 1.0f / rowl[base + r];
    }
  }
  // per-lane V row bases (column d = dt*16+col, elements along S contiguous)
  const _Float16* vbase0 =
      Vt + (((size_t)(b * NHEAD + h)) * DK + col) * S_LEN;
  const _Float16* vbase1 =
      Vt + (((size_t)(b * NHEAD + h)) * DK + 16 + col) * S_LEN;

  v8f ctx0 = {}, ctx1 = {};
  for (int chunk = 0; chunk < S_LEN / 32; ++chunk) {
    for (int i = threadIdx.x; i < 512; i += 256) smean[i] = 0.f;
    __syncthreads();
#pragma unroll
    for (int half = 0; half < 2; ++half) {
      const int s0 = chunk * 32 + half * 16;
      const _Float16* kr = Kh + ((size_t)(s0 + col) * BATCH + b) * D_MODEL + h * DK + hi * 16;
      __builtin_prefetch(kr + (size_t)2 * 16 * BATCH * D_MODEL, 0, 0);
      v16h bf;
#pragma unroll
      for (int e = 0; e < 16; ++e) bf[e] = kr[e];
      v8f c = {};
      c = __builtin_amdgcn_wmma_f32_16x16x32_f16(false, a, false, bf, (short)0, c, false, false);
      const float pks = pk[b * S_LEN + s0 + col];
#pragma unroll
      for (int r = 0; r < 8; ++r) {
        float v = c[r] * SCALE + pqr[r] - pks;
        float p = __expf(v - mrow[r]) * ilrow[r];           // final probability
        atomicAdd(&smean[(r + 8 * hi) * 32 + half * 16 + col], p * 0.125f);
        pstage[wave][(r + 8 * hi) * 32 + half * 16 + col] = (_Float16)p;
      }
    }
    // re-fragment probs as WMMA-A (16 rows x 32 keys) from wave-local LDS
    const _Float16* pr = &pstage[wave][col * 32];
    v16h pa;
#pragma unroll
    for (int e = 0; e < 8; ++e) { pa[e] = pr[hi * 16 + e]; pa[8 + e] = pr[hi * 16 + 8 + e]; }
    // ctx += P·V : B fragments from Vt, 16 contiguous halves per lane
    {
      const int sbase = chunk * 32 + hi * 16;
      v16h vb0, vb1;
#pragma unroll
      for (int e = 0; e < 16; ++e) { vb0[e] = vbase0[sbase + e]; vb1[e] = vbase1[sbase + e]; }
      ctx0 = __builtin_amdgcn_wmma_f32_16x16x32_f16(false, pa, false, vb0, (short)0, ctx0, false, false);
      ctx1 = __builtin_amdgcn_wmma_f32_16x16x32_f16(false, pa, false, vb1, (short)0, ctx1, false, false);
    }
    __syncthreads();
    for (int i = threadIdx.x; i < 512; i += 256) {
      int rr = i >> 5, cc = i & 31;
      meanbuf[((size_t)b * T_LEN + t0 + rr) * S_LEN + chunk * 32 + cc] = smean[i];
    }
    __syncthreads();
  }
  // store ctx (f16) in GEMM row layout: row = t*B+b, col = h*32 + d
#pragma unroll
  for (int r = 0; r < 8; ++r) {
    size_t rowbase = ((size_t)(t0 + r + 8 * hi) * BATCH + b) * D_MODEL + h * DK;
    Ch[rowbase + col]      = (_Float16)ctx0[r];
    Ch[rowbase + 16 + col] = (_Float16)ctx1[r];
  }
}

// ---------------------------------------------------------------------------
// attn_map[b,t,:] = softmax(mean_h attn)[b,t,:]  — one block per row of 1024
// ---------------------------------------------------------------------------
__global__ void __launch_bounds__(256) mean_softmax_kernel(
    const float* __restrict__ meanbuf, float* __restrict__ out) {
  __shared__ float red[256];
  const size_t row = blockIdx.x;
  const float* src = meanbuf + row * S_LEN;
  float vals[4];
#pragma unroll
  for (int i = 0; i < 4; ++i) vals[i] = src[threadIdx.x + i * 256];
  float mx = fmaxf(fmaxf(vals[0], vals[1]), fmaxf(vals[2], vals[3]));
  red[threadIdx.x] = mx;
  __syncthreads();
  for (int s = 128; s > 0; s >>= 1) {
    if (threadIdx.x < s) red[threadIdx.x] = fmaxf(red[threadIdx.x], red[threadIdx.x + s]);
    __syncthreads();
  }
  mx = red[0];
  __syncthreads();
  float sum = 0.f;
#pragma unroll
  for (int i = 0; i < 4; ++i) { vals[i] = __expf(vals[i] - mx); sum += vals[i]; }
  red[threadIdx.x] = sum;
  __syncthreads();
  for (int s = 128; s > 0; s >>= 1) {
    if (threadIdx.x < s) red[threadIdx.x] += red[threadIdx.x + s];
    __syncthreads();
  }
  float inv = 1.0f / red[0];
  float* dst = out + row * S_LEN;
#pragma unroll
  for (int i = 0; i < 4; ++i) dst[threadIdx.x + i * 256] = vals[i] * inv;
}

// ---------------------------------------------------------------------------
extern "C" void kernel_launch(void* const* d_in, const int* in_sizes, int n_in,
                              void* d_out, int out_size, void* d_ws, size_t ws_size,
                              hipStream_t stream) {
  (void)in_sizes; (void)n_in; (void)out_size; (void)ws_size;
  const float* query = (const float*)d_in[0];
  const float* key   = (const float*)d_in[1];
  const float* value = (const float*)d_in[2];
  const float* qxyz  = (const float*)d_in[3];
  const float* kxyz  = (const float*)d_in[4];
  const float* Wq = (const float*)d_in[5];  const float* bq = (const float*)d_in[6];
  const float* Wk = (const float*)d_in[7];  const float* bk = (const float*)d_in[8];
  const float* Wv = (const float*)d_in[9];  const float* bv = (const float*)d_in[10];
  const float* Wo = (const float*)d_in[11]; const float* bo = (const float*)d_in[12];
  const float* pe_w = (const float*)d_in[13];
  const float* pe_b = (const float*)d_in[14];

  // workspace layout (~49 MB)
  _Float16* Qh = (_Float16*)d_ws;
  _Float16* Kh = Qh + (size_t)MROWS * D_MODEL;
  _Float16* Vt = Kh + (size_t)MROWS * D_MODEL;   // [B,H,DK,S] K-major
  _Float16* Ch = Vt + (size_t)MROWS * D_MODEL;
  float* pq      = (float*)(Ch + (size_t)MROWS * D_MODEL);
  float* pk      = pq + BATCH * T_LEN;
  float* rowm    = pk + BATCH * S_LEN;
  float* rowl    = rowm + BATCH * NHEAD * T_LEN;
  float* meanbuf = rowl + BATCH * NHEAD * T_LEN;

  float* out_proj = (float*)d_out;                              // [T,B,D]
  float* attn_map = out_proj + (size_t)T_LEN * BATCH * D_MODEL; // [B,T,S]

  pe_dot_kernel<<<dim3((2 * T_LEN * BATCH + 255) / 256), dim3(256), 0, stream>>>(
      qxyz, kxyz, pe_w, pq, pk);

  dim3 ggrid(MROWS / 16, D_MODEL / 16);
  gemm_f32_to_f16<<<ggrid, dim3(32), 0, stream>>>(query, Wq, bq, Qh);
  gemm_f32_to_f16<<<ggrid, dim3(32), 0, stream>>>(key,   Wk, bk, Kh);
  gemm_f32_to_f16_vt<<<ggrid, dim3(32), 0, stream>>>(value, Wv, bv, Vt);

  attn_pass1<<<dim3(T_LEN / 16, NHEAD, BATCH), dim3(32), 0, stream>>>(
      Qh, Kh, pq, pk, pe_b, rowm, rowl);
  attn_pass2<<<dim3(T_LEN / 16, BATCH), dim3(256), 0, stream>>>(
      Qh, Kh, Vt, pq, pk, pe_b, rowm, rowl, meanbuf, Ch);
  mean_softmax_kernel<<<dim3(BATCH * T_LEN), dim3(256), 0, stream>>>(meanbuf, attn_map);
  gemm_f16_to_f32<<<ggrid, dim3(32), 0, stream>>>(Ch, Wo, bo, out_proj);
}
